// RBFActivation_16149077033818
// MI455X (gfx1250) — matrix operand order repaired and checked by reading, not compiled, not run on hardware
//
#include <hip/hip_runtime.h>

// ---------------------------------------------------------------------------
// RBF activation, MI455X (gfx1250).
//   out = sum_i w[c,i] * exp(-(x - mu_i)^2 / (2 sigma^2)),  K = 31
// Memory roofline: 256 MB @ 23.3 TB/s ~= 11.5 us. Naive = 31 exps/elem
// (trans-bound). Strategy:
//   * geometric-ratio recurrence: 3 exps per lane per 16-pixel tile
//   * v_wmma_f32_16x16x32_f16 (x3, hi/lo split) for the 31-term reduction
//   * TDM tensor_load_to_lds double-buffered staging of x into LDS
// ---------------------------------------------------------------------------

typedef __attribute__((ext_vector_type(16))) _Float16 v16h;
typedef __attribute__((ext_vector_type(8)))  float    v8f;
typedef __attribute__((ext_vector_type(4)))  unsigned v4u;
typedef __attribute__((ext_vector_type(8)))  int      v8i;
typedef __attribute__((ext_vector_type(4)))  int      v4i;

#define KW     31      // number of RBF kernels
#define TILE   2048    // floats per TDM tile (8 KB)
#define CHUNK  16384   // floats per block (8 tiles; divides H*W=65536)
#define NTHREADS 256   // 8 waves of 32

#ifndef HAVE_TDM
#if defined(__has_builtin)
#if __has_builtin(__builtin_amdgcn_tensor_load_to_lds) && \
    __has_builtin(__builtin_amdgcn_s_wait_tensorcnt)
#define HAVE_TDM 1
#endif
#endif
#endif
#ifndef HAVE_TDM
#define HAVE_TDM 0
#endif

__device__ static inline unsigned lds_byte_off(void* p) {
  // LDS aperture: addr[31:0] is the LDS byte offset (ISA 10.2).
  return (unsigned)(unsigned long long)(uintptr_t)p;
}

#if HAVE_TDM
__device__ static inline void tdm_issue_tile(const float* gsrc, unsigned lds_addr) {
  unsigned long long ga = (unsigned long long)(uintptr_t)gsrc;
  // D# group 0: count=1 | lds_addr | global_addr[56:0] | type=2 (bits 127:126)
  v4u g0 = { 1u,
             lds_addr,
             (unsigned)(ga & 0xffffffffull),
             (unsigned)(ga >> 32) | 0x80000000u };
  // D# group 1: 1-row tensor: data_size=4B, tensor_dim0=TILE, tensor_dim1=1,
  // tile_dim0=TILE, tile_dim1=1, dim0_stride=TILE, no multicast/pad/iterate.
  v8i g1 = { 0x00020000,                       // workgroup_mask=0, data_size=2 (4B)
             (int)((TILE & 0xffff) << 16),     // tensor_dim0[15:0] in [31:16]
             (int)((TILE >> 16) | (1 << 16)),  // tensor_dim0[31:16] | tensor_dim1 lo
             (int)(TILE << 16),                // tensor_dim1 hi | tile_dim0
             1,                                // tile_dim1=1, tile_dim2=0
             TILE,                             // tensor_dim0_stride[31:0]
             0, 0 };
  v4i z4 = { 0, 0, 0, 0 };
  v8i z8 = { 0, 0, 0, 0, 0, 0, 0, 0 };
  // clang-23 / therock-10.0 signature: (v4u, v8i, v4i, v4i, v8i, i32 cpol)
  __builtin_amdgcn_tensor_load_to_lds(g0, g1, z4, z4, z8, 0);
}
#endif

__global__ __launch_bounds__(NTHREADS)
void rbf_act_kernel(const float* __restrict__ x,
                    const float* __restrict__ w,
                    const float* __restrict__ mu,
                    const float* __restrict__ sigma,
                    float* __restrict__ out) {
  __shared__ float xs[2][TILE];

  const int tid  = (int)threadIdx.x;
  const int lane = tid & 31;
  const int wid  = tid >> 5;
  const long long chunk = (long long)blockIdx.x * CHUNK;
  const int c = (int)((chunk >> 16) & 63);          // channel: (idx / (H*W)) % C

  // ---- per-dispatch constants (from mu/sigma inputs) --------------------
  const float mu0  = mu[0];
  const float dd   = mu[1] - mu[0];                 // uniform spacing
  const float sg   = sigma[0];
  const float cexp = 1.0f / (2.0f * sg * sg);
  const float A1   = 2.0f * cexp * dd;              // ratio slope
  const float A2   = cexp * dd * dd;                // ratio bias
  const float R    = __expf(-2.0f  * A2);           // ratio-of-ratios
  const float RJ   = __expf(-16.0f * A2);           // jump q_{k+8} -> q_{k+16}

  // ---- B matrix (32x16 f16, all 16 columns = channel weights), hi/lo ----
  // B layout: lane half h, elem e -> K=8h+e (e<8), K=16+8h+(e-8) (e>=8); matches A.
  v16h bhi, blo;
  {
    const int offb = (lane >> 4) << 3;
#pragma unroll
    for (int e = 0; e < 16; ++e) {
      const int k = offb + e + ((e >= 8) ? 8 : 0);
      const float wv = (k < KW) ? w[c * KW + k] : 0.0f;   // pad K=31 with 0
      const _Float16 h = (_Float16)wv;
      bhi[e] = h;
      blo[e] = (_Float16)(wv - (float)h);
    }
  }

  // A layout: lane l holds pixel M=l&15, K in {off..off+7, 16+off..23+off},
  // off = 8*(l>=16)  ->  two runs of 8 consecutive K per lane.
  const int   offa = (lane >> 4) << 3;
  const float muR0 = mu0 + (float)offa * dd;          // run-1 start center
  const float muR1 = mu0 + (float)(16 + offa) * dd;   // run-2 start center

#if HAVE_TDM
  if (wid == 0) tdm_issue_tile(x + chunk, lds_byte_off(&xs[0][0]));
#endif

  const int NT = CHUNK / TILE;
  for (int t = 0; t < NT; ++t) {
    const int buf = t & 1;
#if HAVE_TDM
    if (wid == 0) __builtin_amdgcn_s_wait_tensorcnt(0);   // tile t landed in LDS
    __syncthreads();                                       // publish; t-1 readers done
    if (wid == 0 && (t + 1) < NT)
      tdm_issue_tile(x + chunk + (long long)(t + 1) * TILE,
                     lds_byte_off(&xs[buf ^ 1][0]));
#else
    __syncthreads();
    for (int i = tid; i < TILE; i += NTHREADS)
      xs[buf][i] = x[chunk + (long long)t * TILE + i];
    __syncthreads();
#endif

    // wave `wid` owns 16-pixel groups g = wid, wid+8, ... (128 groups/tile)
#pragma unroll 1
    for (int g = wid; g < TILE / 16; g += 8) {
      const float xv = xs[buf][g * 16 + (lane & 15)];

      v16h ahi, alo;
      // run 1: K = offa .. offa+7
      float d0  = xv - muR0;
      float phi = __expf(-cexp * d0 * d0);
      float q   = __expf(fminf(A1 * d0 - A2, 80.0f));   // clamp: avoid 0*inf
#pragma unroll
      for (int j = 0; j < 8; ++j) {
        const _Float16 h = (_Float16)phi;
        ahi[j] = h;
        alo[j] = (_Float16)(phi - (float)h);
        phi *= q;
        q   *= R;
      }
      // run 2: K = 16+offa .. 23+offa  (q jumps 8 centers: * R^8)
      float d1   = xv - muR1;
      float phi2 = __expf(-cexp * d1 * d1);
      float q2   = q * RJ;
#pragma unroll
      for (int j = 0; j < 8; ++j) {
        const _Float16 h = (_Float16)phi2;
        ahi[8 + j] = h;
        alo[8 + j] = (_Float16)(phi2 - (float)h);
        phi2 *= q2;
        q2   *= R;
      }

      // D = Ahi*Bhi + Alo*Bhi + Ahi*Blo   (f32 accumulate; fp32-grade result)
      v8f acc = {};
      acc = __builtin_amdgcn_wmma_f32_16x16x32_f16(false, ahi, false, bhi,
                                                   (short)0, acc, false, false);
      acc = __builtin_amdgcn_wmma_f32_16x16x32_f16(false, alo, false, bhi,
                                                   (short)0, acc, false, false);
      acc = __builtin_amdgcn_wmma_f32_16x16x32_f16(false, ahi, false, blo,
                                                   (short)0, acc, false, false);

      // D[m,n] = out[m] for every n. Lane 0 holds out[0..7] (VGPR r -> M=r),
      // lane 16 holds out[8..15] (M=8+r). Two b128 stores per active lane.
      if ((lane & 15) == 0) {
        float* op = out + chunk + (long long)t * TILE + g * 16 + ((lane >> 4) << 3);
        *(float4*)(op)     = make_float4(acc[0], acc[1], acc[2], acc[3]);
        *(float4*)(op + 4) = make_float4(acc[4], acc[5], acc[6], acc[7]);
      }
    }
  }
}

extern "C" void kernel_launch(void* const* d_in, const int* in_sizes, int n_in,
                              void* d_out, int out_size, void* d_ws, size_t ws_size,
                              hipStream_t stream) {
  (void)in_sizes; (void)n_in; (void)d_ws; (void)ws_size;
  const float* x     = (const float*)d_in[0];
  const float* w     = (const float*)d_in[1];
  const float* mu    = (const float*)d_in[2];
  const float* sigma = (const float*)d_in[3];
  float* out = (float*)d_out;

  const long long total = (long long)out_size;     // 8*64*256*256
  const int blocks = (int)(total / CHUNK);         // 2048
  rbf_act_kernel<<<dim3(blocks), dim3(NTHREADS), 0, stream>>>(x, w, mu, sigma, out);
}
